// WaveletVortexLayer_22539988369918
// MI455X (gfx1250) — matrix operand order repaired and checked by reading, not compiled, not run on hardware
//
#include <hip/hip_runtime.h>
#include <hip/hip_bf16.h>

// Problem constants (match reference)
#define BATCH   2048
#define NNODES  512
#define HID     64
#define MCHUNK  128          // batch rows per workgroup (8 waves x 16)
#define XPITCH  68           // f32 per X LDS row: 64 + 4 pad (bank spread, 16B aligned)
#define WPITCH  72           // bf16 per W LDS row: 64 + 8 pad (16B-aligned fragments)

typedef __bf16   v16bf __attribute__((ext_vector_type(16)));
typedef __bf16   v8bf  __attribute__((ext_vector_type(8)));
typedef float    v8f   __attribute__((ext_vector_type(8)));
typedef float    v4f   __attribute__((ext_vector_type(4)));
typedef int      v4i_t __attribute__((__vector_size__(16)));  // matches builtin param

union BFrag { v16bf v; v8bf h[2]; };

#if __has_builtin(__builtin_amdgcn_global_load_async_to_lds_b128)
#define XS_ASYNC 1
#endif

static __device__ __forceinline__ void wait_async_zero() {
#if __has_builtin(__builtin_amdgcn_s_wait_asynccnt)
  __builtin_amdgcn_s_wait_asynccnt(0);
#else
  asm volatile("s_wait_asynccnt 0x0" ::: "memory");
#endif
}

__global__ __launch_bounds__(256)
void wavelet_vortex_fused(const float* __restrict__ nf,      // [B,N,H]
                          const float* __restrict__ dec,     // [3,4]
                          const float* __restrict__ rec,     // [3,4]
                          const float* __restrict__ Wg,      // [N,H,H]  (h, k)
                          const float* __restrict__ bg,      // [N,H]
                          const float* __restrict__ gg,      // gamma [N,H]
                          const float* __restrict__ tg,      // beta  [N,H]
                          const int*   __restrict__ scales,  // [N]
                          float* __restrict__ out)           // [B,N,H]
{
  __shared__ __align__(16) float  Xf[MCHUNK * XPITCH];  // X tile, full f32 (34.8 KB)
  __shared__ __align__(16) __bf16 Whi[HID * WPITCH];    // bf16 hi of s*W (9.2 KB)
  __shared__ __align__(16) __bf16 Wlo[HID * WPITCH];    // bf16 lo of s*W (9.2 KB)

  const int n      = blockIdx.x >> 4;            // node id (adjacent blocks share W[n] in L2)
  const int m0     = (blockIdx.x & 15) * MCHUNK; // batch-row base
  const int tid    = threadIdx.x;
  const int lane   = tid & 31;
  const int wid    = tid >> 5;                   // wave id 0..7
  const int l16    = lane & 15;
  const int halfid = lane >> 4;

  // ---- stage X tile -> LDS (f32) via gfx1250 async global->LDS copies -----
  {
    const float* xbase = nf + ((size_t)m0 * NNODES + n) * HID;
    #pragma unroll
    for (int r = 0; r < 8; ++r) {
      int idx = tid + r * 256;               // float4 index; 16 per row
      int row = idx >> 4;
      int c4  = (idx & 15) * 4;
      const float* src = xbase + (size_t)row * (NNODES * HID) + c4;
      float*       dst = &Xf[row * XPITCH + c4];
#ifdef XS_ASYNC
      __builtin_amdgcn_global_load_async_to_lds_b128(
          (v4i_t*)(void*)const_cast<float*>(src),
          (v4i_t*)(void*)dst,
          /*offset=*/0, /*cpol=*/0);
#else
      *(v4f*)dst = *(const v4f*)src;
#endif
    }
  }

  // per-node wavelet scalar: s = dec[sc,1] + rec[sc,1]
  const int   sc = scales[n];
  const float s  = dec[sc * 4 + 1] + rec[sc * 4 + 1];

  // ---- stage s*W[n] -> Whi/Wlo (bf16 hi/lo split), row j = out-chan -------
  {
    const float* wrow = Wg + (size_t)n * (HID * HID);
    #pragma unroll
    for (int r = 0; r < 4; ++r) {
      int idx = tid + r * 256;
      int j   = idx >> 4;
      int c4  = (idx & 15) * 4;
      v4f w = *(const v4f*)(wrow + j * HID + c4);
      #pragma unroll
      for (int i = 0; i < 4; ++i) {
        float  wv = w[i] * s;
        __bf16 wh = (__bf16)wv;
        Whi[j * WPITCH + c4 + i] = wh;
        Wlo[j * WPITCH + c4 + i] = (__bf16)(wv - (float)wh);
      }
    }
  }

  // ---- per-lane epilogue params (col = jt*16 + l16) -----------------------
  float gam[4], bet[4], bia[4];
  #pragma unroll
  for (int jt = 0; jt < 4; ++jt) {
    int col = jt * 16 + l16;
    gam[jt] = gg[n * HID + col];
    bet[jt] = tg[n * HID + col];
    bia[jt] = bg[n * HID + col];
  }

#ifdef XS_ASYNC
  wait_async_zero();   // this wave's async LDS writes landed
#endif
  __syncthreads();     // all waves' staging visible

  // ---- accumulators: C/D layout -> VGPR v holds row (v + 8*halfid), col l16
  // init with bias (constant over rows) => bias folded into GEMM
  v8f acc[4];
  #pragma unroll
  for (int jt = 0; jt < 4; ++jt)
    #pragma unroll
    for (int v = 0; v < 8; ++v)
      acc[jt][v] = bia[jt];

  // ---- split-precision GEMM: T = X @ (s*W)^T, K=64 in two 16x16x32 steps,
  //      3 bf16 WMMAs per step: ah*bh + ah*bl + al*bh (f32 accumulate) ------
  const int arow = (wid * 16 + l16) * XPITCH;
  #pragma unroll
  for (int kt = 0; kt < 2; ++kt) {
    const int kb = kt * 32;

    // A fragment (16x32): lane<16 -> K kb+0..7 & kb+16..23; lane>=16 -> +8
    const int a0 = arow + kb + halfid * 8;
    v4f x0 = *(const v4f*)(&Xf[a0]);
    v4f x1 = *(const v4f*)(&Xf[a0 + 4]);
    v4f x2 = *(const v4f*)(&Xf[a0 + 16]);
    v4f x3 = *(const v4f*)(&Xf[a0 + 20]);
    float xs[16];
    #pragma unroll
    for (int i = 0; i < 4; ++i) {
      xs[i]      = x0[i];
      xs[4 + i]  = x1[i];
      xs[8 + i]  = x2[i];
      xs[12 + i] = x3[i];
    }
    v16bf ah, al;
    #pragma unroll
    for (int i = 0; i < 16; ++i) {
      __bf16 h = (__bf16)xs[i];
      ah[i] = h;
      al[i] = (__bf16)(xs[i] - (float)h);
    }

    // hoist all B fragments of this k-step so ds_loads issue back-to-back
    BFrag bh[4], bl[4];
    #pragma unroll
    for (int jt = 0; jt < 4; ++jt) {
      const int off = (jt * 16 + l16) * WPITCH + kb + halfid * 8;
      bh[jt].h[0] = *(const v8bf*)(&Whi[off]);
      bh[jt].h[1] = *(const v8bf*)(&Whi[off + 16]);
      bl[jt].h[0] = *(const v8bf*)(&Wlo[off]);
      bl[jt].h[1] = *(const v8bf*)(&Wlo[off + 16]);
    }

    #pragma unroll
    for (int jt = 0; jt < 4; ++jt) {
      acc[jt] = __builtin_amdgcn_wmma_f32_16x16x32_bf16(
          false, al, false, bh[jt].v, (short)0, acc[jt], false, false);
      acc[jt] = __builtin_amdgcn_wmma_f32_16x16x32_bf16(
          false, ah, false, bl[jt].v, (short)0, acc[jt], false, false);
      acc[jt] = __builtin_amdgcn_wmma_f32_16x16x32_bf16(
          false, ah, false, bh[jt].v, (short)0, acc[jt], false, false);
    }
  }

  // ---- fused LayerNorm (biased var, eps=1e-5) + leaky ReLU + store --------
  #pragma unroll
  for (int v = 0; v < 8; ++v) {
    float s1 = acc[0][v] + acc[1][v] + acc[2][v] + acc[3][v];
    float s2 = acc[0][v] * acc[0][v] + acc[1][v] * acc[1][v] +
               acc[2][v] * acc[2][v] + acc[3][v] * acc[3][v];
    // reduce over the 16 lanes of this half-wave (row owners); xor masks
    // 1,2,4,8 never cross the half boundary
    #pragma unroll
    for (int off = 1; off < 16; off <<= 1) {
      s1 += __shfl_xor(s1, off, 32);
      s2 += __shfl_xor(s2, off, 32);
    }
    const float mu  = s1 * (1.0f / 64.0f);
    const float var = s2 * (1.0f / 64.0f) - mu * mu;
    const float inv = __frsqrt_rn(var + 1e-5f);

    const int row = m0 + wid * 16 + v + halfid * 8;
    float* orow = out + ((size_t)row * NNODES + n) * HID;
    #pragma unroll
    for (int jt = 0; jt < 4; ++jt) {
      float o = (acc[jt][v] - mu) * inv * gam[jt] + bet[jt];
      o = (o >= 0.0f) ? o : 0.01f * o;
      orow[jt * 16 + l16] = o;
    }
  }
}

extern "C" void kernel_launch(void* const* d_in, const int* in_sizes, int n_in,
                              void* d_out, int out_size, void* d_ws, size_t ws_size,
                              hipStream_t stream) {
  const float* nf     = (const float*)d_in[0];
  const float* dec    = (const float*)d_in[1];
  const float* rec    = (const float*)d_in[2];
  const float* W      = (const float*)d_in[3];
  const float* b      = (const float*)d_in[4];
  const float* gamma  = (const float*)d_in[5];
  const float* beta   = (const float*)d_in[6];
  const int*   scales = (const int*)d_in[7];
  float* out = (float*)d_out;

  dim3 grid(NNODES * (BATCH / MCHUNK)); // 512 * 16 = 8192
  dim3 block(256);                      // 8 wave32
  wavelet_vortex_fused<<<grid, block, 0, stream>>>(
      nf, dec, rec, W, b, gamma, beta, scales, out);
}
